// Grapher_68719476754
// MI455X (gfx1250) — compile-verified, matrix operand order checked by reading, not run on hardware
//
#include <hip/hip_runtime.h>
#include <math.h>

typedef __attribute__((ext_vector_type(16))) __bf16        v16bf;
typedef __attribute__((ext_vector_type(8)))  float         v8f;
typedef __attribute__((ext_vector_type(8)))  unsigned int  v8u;
typedef __attribute__((ext_vector_type(4)))  unsigned int  v4u;

#define BM 64
#define BN 64
#define BK 32        // K-block = one bf16 WMMA K
#define LDP 20       // LDS row pad (dwords): 20*j mod 64 distinct & 4-aligned -> conflict-free b128
#define HWPIX 3136   // 56*56 pixels per batch image
#define EPSBN 1e-5f

// ---- fp32 -> split bf16 (hi+lo), RNE both times; hi+lo reproduces ~16 mantissa bits ----
__device__ __forceinline__ unsigned f2bf_rne(float f) {
    unsigned u = __float_as_uint(f);
    return (u + 0x7fffu + ((u >> 16) & 1u)) >> 16;
}
__device__ __forceinline__ float bf2f(unsigned h) { return __uint_as_float(h << 16); }
__device__ __forceinline__ uint2 split2(float a0, float a1) {
    unsigned h0 = f2bf_rne(a0), h1 = f2bf_rne(a1);
    float r0 = a0 - bf2f(h0), r1 = a1 - bf2f(h1);
    unsigned l0 = f2bf_rne(r0), l1 = f2bf_rne(r1);
    uint2 r; r.x = h0 | (h1 << 16); r.y = l0 | (l1 << 16); return r;
}

// Fragment gather: rows-of-16-kpairs layout -> two contiguous b128 loads per matrix.
// VGPR v <-> kpair v + 4*lhi + (v>=4 ? 4 : 0)  ==> quads [4*lhi .. +3] and [8+4*lhi .. +3]
__device__ __forceinline__ v16bf frag_ld(const unsigned* rowptr, int lhi) {
    v4u q0 = *(const v4u*)(rowptr + 4 * lhi);
    v4u q1 = *(const v4u*)(rowptr + 8 + 4 * lhi);
    v8u u  = __builtin_shufflevector(q0, q1, 0, 1, 2, 3, 4, 5, 6, 7);
    return __builtin_bit_cast(v16bf, u);
}

// out[p, o] = act( bn( sum_k A[p,k] * W[o,k] + cb[o] ) )  (+ residual for mode 1)
// mode 0: pixel-major store, no activation
// mode 1: NCHW store + residual add (final layer)
// mode 2: pixel-major store + exact GELU
// GEMM is split-bf16: acc += Ahi*Blo + Alo*Bhi + Ahi*Bhi  (error ~2^-16 relative)
__global__ __launch_bounds__(128) void gemm_bn_act_kernel(
    const float* __restrict__ A, long a_batch, long a_p, long a_k,
    const float* __restrict__ W, int Cin, int Cout,
    const float* __restrict__ cb, const float* __restrict__ gam,
    const float* __restrict__ bet, const float* __restrict__ mu,
    const float* __restrict__ var,
    float* __restrict__ Out, long o_batch, long o_p, int o_coloff,
    const float* __restrict__ Res,
    int mode, int mblocks_per_batch)
{
    // de-interleaved tiles: [row][kpair] bf16-pair dwords; hi and lo in separate arrays
    __shared__ unsigned lAhi[BM][LDP], lAlo[BM][LDP];   // A tile: [m][k/2]
    __shared__ unsigned lWhi[BN][LDP], lWlo[BN][LDP];   // W tile: [n][k/2]

    const int bm    = blockIdx.x % mblocks_per_batch;
    const int batch = blockIdx.x / mblocks_per_batch;
    const int nb0   = blockIdx.y * BN;
    const int p0    = bm * BM;

    const int t    = threadIdx.x;
    const int wave = t >> 5;
    const int lane = t & 31;
    const int lrow = lane & 15;   // M (A) / N (B,C) index within 16
    const int lhi  = lane >> 4;   // half-wave select

    v8f acc[4] = {};              // 4 N-subtiles of 16x16 f32

    const float* Ab = A + (long)batch * a_batch;

    for (int k0 = 0; k0 < Cin; k0 += BK) {
        // ---- stage + convert A tile [BM x BK] ----
        if (a_k != 1) {
            // k-major memory (NCHW x): contiguous along pixels -> lanes sweep m
            #pragma unroll
            for (int i = 0; i < 8; ++i) {
                int e  = i * 128 + t;
                int kp = e >> 6, mm = e & 63;
                const float* s = Ab + (long)(p0 + mm) * a_p + (long)(k0 + 2 * kp) * a_k;
                uint2 d = split2(s[0], s[a_k]);
                lAhi[mm][kp] = d.x; lAlo[mm][kp] = d.y;
            }
        } else {
            // pixel-major memory (y/g): contiguous along channels -> lanes sweep k
            #pragma unroll
            for (int i = 0; i < 8; ++i) {
                int e  = i * 128 + t;
                int mm = e >> 4, kp = e & 15;
                const float* s = Ab + (long)(p0 + mm) * a_p + (k0 + 2 * kp);
                uint2 d = split2(s[0], s[1]);
                lAhi[mm][kp] = d.x; lAlo[mm][kp] = d.y;
            }
        }
        // ---- stage + convert W tile [BN x BK]  (w is [Cout][Cin] row-major) ----
        #pragma unroll
        for (int i = 0; i < 8; ++i) {
            int e  = i * 128 + t;
            int n  = e >> 4, kp = e & 15;
            const float* s = W + (long)(nb0 + n) * Cin + (k0 + 2 * kp);
            uint2 d = split2(s[0], s[1]);
            lWhi[n][kp] = d.x; lWlo[n][kp] = d.y;
        }
        // prefetch next K tile (global_prefetch_b8)
        if (k0 + BK < Cin) {
            __builtin_prefetch(W + (long)(nb0 + (t >> 4)) * Cin + (k0 + BK + 2 * (t & 15)), 0, 1);
            __builtin_prefetch(Ab + (long)(p0 + (t & 63)) * a_p +
                               (long)(k0 + BK + 2 * (t >> 6)) * a_k, 0, 1);
        }
        __syncthreads();

        // ---- compute: wave handles M rows [wave*16, wave*16+16) ----
        const int arow = wave * 16 + lrow;
        const v16bf Ahi = frag_ld(&lAhi[arow][0], lhi);
        const v16bf Alo = frag_ld(&lAlo[arow][0], lhi);

        #pragma unroll
        for (int nt = 0; nt < 4; ++nt) {
            const int nrow = nt * 16 + lrow;
            const v16bf Bhi = frag_ld(&lWhi[nrow][0], lhi);
            const v16bf Blo = frag_ld(&lWlo[nrow][0], lhi);
            // small cross terms first, dominant term last
            acc[nt] = __builtin_amdgcn_wmma_f32_16x16x32_bf16(
                false, Ahi, false, Blo, (short)0, acc[nt], false, false);
            acc[nt] = __builtin_amdgcn_wmma_f32_16x16x32_bf16(
                false, Alo, false, Bhi, (short)0, acc[nt], false, false);
            acc[nt] = __builtin_amdgcn_wmma_f32_16x16x32_bf16(
                false, Ahi, false, Bhi, (short)0, acc[nt], false, false);
        }
        __syncthreads();
    }

    // ---- epilogue: fused conv-bias + BN (+gelu / +residual) ----
    #pragma unroll
    for (int nt = 0; nt < 4; ++nt) {
        const int ch = nb0 + nt * 16 + lrow;           // C-tile: N = lane%16 (constant/lane)
        const float inv = gam[ch] / sqrtf(var[ch] + EPSBN);
        const float sh  = (cb[ch] - mu[ch]) * inv + bet[ch];
        #pragma unroll
        for (int r = 0; r < 8; ++r) {
            float val = acc[nt][r] * inv + sh;
            if (mode == 2)
                val = 0.5f * val * (1.0f + erff(val * 0.70710678118654752f));
            const int mm = wave * 16 + r + 8 * lhi;    // C-tile: M = r + 8*(l/16)
            const int p  = p0 + mm;
            if (mode == 1) {
                const long addr = ((long)batch * Cout + ch) * HWPIX + p;
                Out[addr] = val + Res[addr];
            } else {
                Out[(long)batch * o_batch + (long)p * o_p + (long)(o_coloff + ch)] = val;
            }
        }
    }
}

// x_j = max(0, h - min over 27 same-parity H-shifts and 27 same-parity W-shifts)
// y layout: [B][H][W][768]; h1 in channels [0,384), x_j written to [384,768)
__global__ void mrconv_xj_kernel(float* __restrict__ y)
{
    const long total = 16L * 56 * 56 * 384;
    long t = (long)blockIdx.x * blockDim.x + threadIdx.x;
    if (t >= total) return;
    const int c  = (int)(t % 384);
    long pw      = t / 384;
    const int w  = (int)(pw % 56);
    long bh      = pw / 56;
    const int h  = (int)(bh % 56);
    const int b  = (int)(bh / 56);

    const long base = ((long)((b * 56 + h) * 56 + w)) * 768;
    const float self = y[base + c];
    float mn = 3.402823466e38f;
    #pragma unroll 1
    for (int i = 2; i < 56; i += 2) {
        int hp = h - i; if (hp < 0) hp += 56;
        mn = fminf(mn, y[((long)((b * 56 + hp) * 56 + w)) * 768 + c]);
        int wp = w - i; if (wp < 0) wp += 56;
        mn = fminf(mn, y[((long)((b * 56 + h) * 56 + wp)) * 768 + c]);
    }
    y[base + 384 + c] = fmaxf(0.0f, self - mn);
}

extern "C" void kernel_launch(void* const* d_in, const int* in_sizes, int n_in,
                              void* d_out, int out_size, void* d_ws, size_t ws_size,
                              hipStream_t stream)
{
    const float* x      = (const float*)d_in[0];
    const float* fc1_w  = (const float*)d_in[1];
    const float* fc1_b  = (const float*)d_in[2];
    const float* fc1_g  = (const float*)d_in[3];
    const float* fc1_be = (const float*)d_in[4];
    const float* fc1_m  = (const float*)d_in[5];
    const float* fc1_v  = (const float*)d_in[6];
    const float* gc_w   = (const float*)d_in[7];
    const float* gc_b   = (const float*)d_in[8];
    const float* gc_g   = (const float*)d_in[9];
    const float* gc_be  = (const float*)d_in[10];
    const float* gc_m   = (const float*)d_in[11];
    const float* gc_v   = (const float*)d_in[12];
    const float* fc2_w  = (const float*)d_in[13];
    const float* fc2_b  = (const float*)d_in[14];
    const float* fc2_g  = (const float*)d_in[15];
    const float* fc2_be = (const float*)d_in[16];
    const float* fc2_m  = (const float*)d_in[17];
    const float* fc2_v  = (const float*)d_in[18];
    (void)in_sizes; (void)n_in; (void)out_size; (void)ws_size;

    // workspace: y [50176 x 768] then g [50176 x 768]  (2 x 154 MB)
    float* y = (float*)d_ws;
    float* g = y + 50176L * 768;

    const int mpb = HWPIX / BM;           // 49 M-blocks per batch image
    dim3 blk(128);

    // GEMM1: h = bn(x @ fc1_wT + b) -> y[:, 0:384]
    gemm_bn_act_kernel<<<dim3(mpb * 16, 384 / BN), blk, 0, stream>>>(
        x, 384L * HWPIX, 1L, (long)HWPIX,
        fc1_w, 384, 384, fc1_b, fc1_g, fc1_be, fc1_m, fc1_v,
        y, 768L * HWPIX, 768L, 0, nullptr, 0, mpb);

    // mrconv: y[:, 384:768] = x_j
    mrconv_xj_kernel<<<(16L * 56 * 56 * 384 + 255) / 256, 256, 0, stream>>>(y);

    // GEMM2: g = gelu(bn(y @ gc_wT + b))
    gemm_bn_act_kernel<<<dim3(mpb * 16, 768 / BN), blk, 0, stream>>>(
        y, 768L * HWPIX, 768L, 1L,
        gc_w, 768, 768, gc_b, gc_g, gc_be, gc_m, gc_v,
        g, 768L * HWPIX, 768L, 0, nullptr, 2, mpb);

    // GEMM3: out = bn(g @ fc2_wT + b) + x   (NCHW store + residual)
    gemm_bn_act_kernel<<<dim3(mpb * 16, 384 / BN), blk, 0, stream>>>(
        g, 768L * HWPIX, 768L, 1L,
        fc2_w, 768, 384, fc2_b, fc2_g, fc2_be, fc2_m, fc2_v,
        (float*)d_out, 0L, 0L, 0, x, 1, mpb);
}
